// _GRUNet_6871947674314
// MI455X (gfx1250) — matrix operand order repaired and checked by reading, not compiled, not run on hardware
//
#include <hip/hip_runtime.h>
#include <hip/hip_bf16.h>

// ---------------- problem constants ----------------
#define BATCH 4096
#define TT    256
#define II    6
#define HH    64
#define GG    192            // 3*H
#define ROWS_PER_WAVE 16
#define WSTR  68             // LDS stride (ushorts) for 64-wide weight rows (136B, 8B aligned)
#define TSTR  68             // LDS stride (ushorts) for 64-wide h tiles

typedef __attribute__((ext_vector_type(16))) __bf16 v16bf;
typedef __attribute__((ext_vector_type(8)))  float  v8f;

__device__ __forceinline__ unsigned short f2bf(float f) {
    union { float f; unsigned int u; } v; v.f = f;
    unsigned int u = v.u;
    u += 0x7FFFu + ((u >> 16) & 1u);          // round-to-nearest-even
    return (unsigned short)(u >> 16);
}
__device__ __forceinline__ unsigned int pack2(unsigned short lo, unsigned short hi) {
    return (unsigned int)lo | ((unsigned int)hi << 16);
}
// fast activations: v_exp_f32 + v_rcp_f32 (no IEEE div expansion)
__device__ __forceinline__ float sigm(float v) {
    return __builtin_amdgcn_rcpf(1.0f + __expf(-v));
}
__device__ __forceinline__ float tanh_(float v) {
    return 2.0f * __builtin_amdgcn_rcpf(1.0f + __expf(-2.0f * v)) - 1.0f;
}

__device__ __forceinline__ v8f wmma_bf16(v16bf a, v16bf b, v8f c) {
    return __builtin_amdgcn_wmma_f32_16x16x32_bf16(false, a, false, b, (short)0, c, false, false);
}

union FragU { uint2 q[4]; unsigned int u[8]; v16bf v; };

// A-fragment (16x32 bf16) from an LDS tile stored row-major [16][TSTR]
// regs 0..3 hold k = (half*8+kc)..+7 ; regs 4..7 hold k = 16+(half*8+kc)..+7
__device__ __forceinline__ v16bf load_a_frag(const unsigned short* tile, int lane, int kc) {
    FragU f;
    const int m    = lane & 15;
    const int half = (lane >> 4) & 1;
    const unsigned short* row = tile + m * TSTR;
    const int kb = half * 8 + kc;
    f.q[0] = *(const uint2*)(row + kb);
    f.q[1] = *(const uint2*)(row + kb + 4);
    f.q[2] = *(const uint2*)(row + kb + 16);
    f.q[3] = *(const uint2*)(row + kb + 20);
    return f.v;
}

// B-fragment (32x16 bf16) from LDS weights stored row-major [192][WSTR] (row = out index n)
// regs 0..3 hold k = kbase..+7 ; regs 4..7 hold k = kbase+16..+7 (kbase = kc + half*8)
__device__ __forceinline__ v16bf load_b_frag(const unsigned short* w, int lane, int jtile, int kc) {
    FragU f;
    const int n     = jtile * 16 + (lane & 15);
    const int kbase = kc + ((lane >> 4) & 1) * 8;
    const unsigned short* row = w + n * WSTR + kbase;
    f.q[0] = *(const uint2*)(row);
    f.q[1] = *(const uint2*)(row + 4);
    f.q[2] = *(const uint2*)(row + 16);
    f.q[3] = *(const uint2*)(row + 20);
    return f.v;
}

// B-fragment for W_ih0 : [192][8] bf16, only k<6 nonzero (zero-padded K to 32)
__device__ __forceinline__ v16bf load_b_frag_ih0(const unsigned short* w, int lane, int jtile) {
    FragU f;
    const int n = jtile * 16 + (lane & 15);
    const unsigned short* row = w + n * 8;
    uint2 z2; z2.x = 0u; z2.y = 0u;
    if (lane < 16) {
        f.q[0] = *(const uint2*)(row);
        f.q[1] = *(const uint2*)(row + 4);
    } else {
        f.q[0] = z2; f.q[1] = z2;
    }
    f.q[2] = z2; f.q[3] = z2;
    return f.v;
}

__device__ __forceinline__ v8f splat8(float s) {
    v8f c;
#pragma unroll
    for (int i = 0; i < 8; ++i) c[i] = s;
    return c;
}

// One fused GRU cell step for a 16-row tile.
// XCHUNKS==1 -> layer 0 (input K padded 6->32, special W_ih); XCHUNKS==2 -> layer 1 (input K=64).
// Biases are register-resident per lane (hoisted out of the recurrence).
template <int XCHUNKS>
__device__ __forceinline__ void gru_step(v8f (&hC)[4], v16bf xA0, v16bf xA1,
                                         const unsigned short* htile, unsigned short* out_tile,
                                         const unsigned short* w_ih, const unsigned short* w_hh,
                                         const float (&b_r)[4], const float (&b_z)[4],
                                         const float (&b_xn)[4], const float (&b_hn)[4],
                                         int lane) {
    const int nl   = lane & 15;
    const int half = (lane >> 4) & 1;
    v16bf hA0 = load_a_frag(htile, lane, 0);
    v16bf hA1 = load_a_frag(htile, lane, 32);

    v8f rv[4];
    // ---- r gate (weight rows 0..63) ----
#pragma unroll
    for (int j = 0; j < 4; ++j) {
        v8f c = splat8(b_r[j]);
        if (XCHUNKS == 1) {
            c = wmma_bf16(xA0, load_b_frag_ih0(w_ih, lane, j), c);
        } else {
            c = wmma_bf16(xA0, load_b_frag(w_ih, lane, j, 0), c);
            c = wmma_bf16(xA1, load_b_frag(w_ih, lane, j, 32), c);
        }
        c = wmma_bf16(hA0, load_b_frag(w_hh, lane, j, 0), c);
        c = wmma_bf16(hA1, load_b_frag(w_hh, lane, j, 32), c);
#pragma unroll
        for (int i = 0; i < 8; ++i) rv[j][i] = sigm(c[i]);
    }
    // ---- hn (weight rows 128..191 of W_hh), fold r*hn into rv ----
#pragma unroll
    for (int j = 0; j < 4; ++j) {
        v8f c = splat8(b_hn[j]);
        c = wmma_bf16(hA0, load_b_frag(w_hh, lane, 8 + j, 0), c);
        c = wmma_bf16(hA1, load_b_frag(w_hh, lane, 8 + j, 32), c);
#pragma unroll
        for (int i = 0; i < 8; ++i) rv[j][i] *= c[i];
    }
    // ---- xn (weight rows 128..191 of W_ih); n = tanh(xn + r*hn) ----
#pragma unroll
    for (int j = 0; j < 4; ++j) {
        v8f c = splat8(b_xn[j]);
        if (XCHUNKS == 1) {
            c = wmma_bf16(xA0, load_b_frag_ih0(w_ih, lane, 8 + j), c);
        } else {
            c = wmma_bf16(xA0, load_b_frag(w_ih, lane, 8 + j, 0), c);
            c = wmma_bf16(xA1, load_b_frag(w_ih, lane, 8 + j, 32), c);
        }
#pragma unroll
        for (int i = 0; i < 8; ++i) rv[j][i] = tanh_(c[i] + rv[j][i]);
    }
    // ---- z gate (weight rows 64..127); h = (1-z)*n + z*h ----
#pragma unroll
    for (int j = 0; j < 4; ++j) {
        v8f c = splat8(b_z[j]);
        if (XCHUNKS == 1) {
            c = wmma_bf16(xA0, load_b_frag_ih0(w_ih, lane, 4 + j), c);
        } else {
            c = wmma_bf16(xA0, load_b_frag(w_ih, lane, 4 + j, 0), c);
            c = wmma_bf16(xA1, load_b_frag(w_ih, lane, 4 + j, 32), c);
        }
        c = wmma_bf16(hA0, load_b_frag(w_hh, lane, 4 + j, 0), c);
        c = wmma_bf16(hA1, load_b_frag(w_hh, lane, 4 + j, 32), c);
#pragma unroll
        for (int i = 0; i < 8; ++i) {
            float z = sigm(c[i]);
            hC[j][i] = (1.0f - z) * rv[j][i] + z * hC[j][i];
        }
    }
    // ---- write new h (bf16) to LDS tile for the next consumer ----
#pragma unroll
    for (int j = 0; j < 4; ++j)
#pragma unroll
        for (int rr = 0; rr < 8; ++rr)
            out_tile[(rr + half * 8) * TSTR + j * 16 + nl] = f2bf(hC[j][rr]);
}

__global__ __launch_bounds__(32) void gru_fused_kernel(
    const float* __restrict__ x,
    const float* __restrict__ W_ih0, const float* __restrict__ W_hh0,
    const float* __restrict__ b_ih0, const float* __restrict__ b_hh0,
    const float* __restrict__ W_ih1, const float* __restrict__ W_hh1,
    const float* __restrict__ b_ih1, const float* __restrict__ b_hh1,
    const float* __restrict__ W_fc,  const float* __restrict__ b_fc,
    float* __restrict__ out) {
    extern __shared__ char smem[];
    unsigned short* lw_hh0 = (unsigned short*)smem;                       // [192][WSTR]
    unsigned short* lw_ih1 = lw_hh0 + GG * WSTR;
    unsigned short* lw_hh1 = lw_ih1 + GG * WSTR;
    unsigned short* lw_ih0 = lw_hh1 + GG * WSTR;                          // [192][8]
    unsigned short* tile01 = lw_ih0 + GG * 8;                             // [16][TSTR]
    unsigned short* tile2  = tile01 + 16 * TSTR;                          // [16][TSTR]

    const int lane = threadIdx.x & 31;
    const int nl   = lane & 15;
    const int half = (lane >> 4) & 1;
    const int b0   = blockIdx.x * ROWS_PER_WAVE;   // one 16-row wave per workgroup/WGP

    // ---- weight prep: fp32 global -> bf16 LDS (padded strides) ----
    for (int idx = lane; idx < GG * HH; idx += 32) {
        int g = idx >> 6, k = idx & 63;
        lw_hh0[g * WSTR + k] = f2bf(W_hh0[idx]);
        lw_ih1[g * WSTR + k] = f2bf(W_ih1[idx]);
        lw_hh1[g * WSTR + k] = f2bf(W_hh1[idx]);
    }
    for (int idx = lane; idx < GG * 8; idx += 32) {
        int g = idx >> 3, k = idx & 7;
        lw_ih0[idx] = (k < II) ? f2bf(W_ih0[g * II + k]) : (unsigned short)0;
    }
    for (int idx = lane; idx < 2 * 16 * TSTR; idx += 32) tile01[idx] = 0;
    __syncthreads();

    // ---- biases: register-resident per lane (column n = j*16+nl) ----
    float br0[4], bz0[4], bxn0[4], bhn0[4];
    float br1[4], bz1[4], bxn1[4], bhn1[4];
#pragma unroll
    for (int j = 0; j < 4; ++j) {
        const int n = j * 16 + nl;
        br0[j]  = b_ih0[n]       + b_hh0[n];
        bz0[j]  = b_ih0[64 + n]  + b_hh0[64 + n];
        bxn0[j] = b_ih0[128 + n];
        bhn0[j] = b_hh0[128 + n];
        br1[j]  = b_ih1[n]       + b_hh1[n];
        bz1[j]  = b_ih1[64 + n]  + b_hh1[64 + n];
        bxn1[j] = b_ih1[128 + n];
        bhn1[j] = b_hh1[128 + n];
    }

    // ---- per-wave hidden states in WMMA C-layout ----
    v8f hC0[4], hC2[4];
#pragma unroll
    for (int j = 0; j < 4; ++j) { hC0[j] = splat8(0.0f); hC2[j] = splat8(0.0f); }

    FragU zf;
#pragma unroll
    for (int r = 0; r < 8; ++r) zf.u[r] = 0u;
    const v16bf zeroFrag = zf.v;

    const float* xrow = x + (size_t)(b0 + nl) * TT * II;

    // ---- recurrence over T; both layers fused per step ----
#pragma unroll 1
    for (int t = 0; t < TT; ++t) {
        // x_t A-fragment (rows m=0..15 in lanes 0..15, K 0..5 nonzero)
        FragU xf;
#pragma unroll
        for (int r = 0; r < 8; ++r) xf.u[r] = 0u;
        if (lane < 16) {
            const float* p = xrow + t * II;
            xf.u[0] = pack2(f2bf(p[0]), f2bf(p[1]));
            xf.u[1] = pack2(f2bf(p[2]), f2bf(p[3]));
            xf.u[2] = pack2(f2bf(p[4]), f2bf(p[5]));
        }
        gru_step<1>(hC0, xf.v, zeroFrag, tile01, tile01,
                    lw_ih0, lw_hh0, br0, bz0, bxn0, bhn0, lane);

        v16bf x1A0 = load_a_frag(tile01, lane, 0);
        v16bf x1A1 = load_a_frag(tile01, lane, 32);
        gru_step<2>(hC2, x1A0, x1A1, tile2, tile2,
                    lw_ih1, lw_hh1, br1, bz1, bxn1, bhn1, lane);
    }

    // ---- final FC: out[b] = h2_last . W_fc + b_fc ----
    float wv[4];
#pragma unroll
    for (int j = 0; j < 4; ++j) wv[j] = W_fc[j * 16 + nl];
    const float bfc = b_fc[0];
#pragma unroll
    for (int rr = 0; rr < 8; ++rr) {
        float p = 0.0f;
#pragma unroll
        for (int j = 0; j < 4; ++j) p += hC2[j][rr] * wv[j];
#pragma unroll
        for (int m = 1; m < 16; m <<= 1) p += __shfl_xor(p, m, 32);
        if (nl == 0) out[b0 + rr + half * 8] = p + bfc;
    }
}

extern "C" void kernel_launch(void* const* d_in, const int* in_sizes, int n_in,
                              void* d_out, int out_size, void* d_ws, size_t ws_size,
                              hipStream_t stream) {
    (void)in_sizes; (void)n_in; (void)d_ws; (void)ws_size; (void)out_size;
    const float* x     = (const float*)d_in[0];
    const float* W_ih0 = (const float*)d_in[1];
    const float* W_hh0 = (const float*)d_in[2];
    const float* b_ih0 = (const float*)d_in[3];
    const float* b_hh0 = (const float*)d_in[4];
    const float* W_ih1 = (const float*)d_in[5];
    const float* W_hh1 = (const float*)d_in[6];
    const float* b_ih1 = (const float*)d_in[7];
    const float* b_hh1 = (const float*)d_in[8];
    const float* W_fc  = (const float*)d_in[9];
    const float* b_fc  = (const float*)d_in[10];
    float* out = (float*)d_out;

    const int smem_bytes = 3 * GG * WSTR * 2      // 3 bf16 64-wide weight mats
                         + GG * 8 * 2             // padded W_ih0
                         + 2 * 16 * TSTR * 2;     // h tiles (1 wave)
    hipFuncSetAttribute((const void*)gru_fused_kernel,
                        hipFuncAttributeMaxDynamicSharedMemorySize, smem_bytes);

    dim3 grid(BATCH / ROWS_PER_WAVE);   // 256 workgroups -> one recurrence wave per WGP
    dim3 block(32);                     // single wave32
    gru_fused_kernel<<<grid, block, smem_bytes, stream>>>(
        x, W_ih0, W_hh0, b_ih0, b_hh0, W_ih1, W_hh1, b_ih1, b_hh1, W_fc, b_fc, out);
}
